// HiTop_52329881534822
// MI455X (gfx1250) — matrix operand order repaired and checked by reading, not compiled, not run on hardware
//
#include <hip/hip_runtime.h>
#include <hip/hip_bf16.h>

// ---------------------------------------------------------------------------
// Hierarchical SoftSOM autoencoder forward for MI455X (gfx1250, wave32).
// All conv / FC / SOM matmuls: v_wmma_f32_16x16x32_f16, f32 accumulate.
// f16 operands (converted once), 16x64 register-blocked wave tiles,
// double-buffered B tiles staged with GLOBAL_LOAD_ASYNC_TO_LDS_B128
// (ASYNCcnt-tracked DMA overlapping the WMMAs), b128 loads, prefetch.
// ---------------------------------------------------------------------------

typedef __attribute__((ext_vector_type(16))) _Float16 v16h;
typedef __attribute__((ext_vector_type(8)))  _Float16 h8;
typedef __attribute__((ext_vector_type(8)))  float    v8f;

#define BATCH 2048
#define DLAT  256
#define INV_TEMP 3.3333333333333335f   // 1/0.3

// ---------------- CDNA5 async global->LDS copy (ASYNCcnt) ------------------
// LDS byte offset = low 32 bits of the generic pointer (flat LDS aperture).
__device__ __forceinline__ void async_copy_b128(const _Float16* g, _Float16* l) {
  unsigned int lo = (unsigned int)(uintptr_t)l;
  asm volatile("global_load_async_to_lds_b128 %0, %1, off"
               :: "v"(lo), "v"(g) : "memory");
}
__device__ __forceinline__ void wait_async0() {
  asm volatile("s_wait_asynccnt 0x0" ::: "memory");
}

// ---------------- f32 -> f16 conversion with K-padding ---------------------
__global__ void cvt_pad_kernel(const float* __restrict__ src,
                               _Float16* __restrict__ dst,
                               int K, int KP, size_t total)
{
  size_t i = (size_t)blockIdx.x * 256 + threadIdx.x;
  if (i >= total) return;
  size_t r = i / (size_t)KP;
  int k = (int)(i - r * (size_t)KP);
  dst[i] = (_Float16)((k < K) ? src[r * (size_t)K + k] : 0.0f);
}

// protosT: dst[d*KP + j] = (j < Kn) ? (f16)src[j*256 + d] : 0
__global__ void transpose_pad_kernel(const float* __restrict__ src,
                                     _Float16* __restrict__ dst,
                                     int Kn, int KP)
{
  size_t i = (size_t)blockIdx.x * 256 + threadIdx.x;
  if (i >= (size_t)DLAT * KP) return;
  int d = (int)(i / (size_t)KP);
  int j = (int)(i - (size_t)d * KP);
  dst[i] = (_Float16)((j < Kn) ? src[(size_t)j * DLAT + d] : 0.0f);
}

// ---------------- bilinear 2x upsample fetch on f16 planes -----------------
__device__ __forceinline__ float up2_fetch(const _Float16* __restrict__ p,
                                           int H, int W, int iy, int ix) {
  float sy = 0.5f * (float)iy - 0.25f;
  float fy = floorf(sy);  float ty = sy - fy;
  int y0 = (int)fy, y1 = y0 + 1;
  y0 = y0 < 0 ? 0 : (y0 >= H ? H - 1 : y0);
  y1 = y1 < 0 ? 0 : (y1 >= H ? H - 1 : y1);
  float sx = 0.5f * (float)ix - 0.25f;
  float fx = floorf(sx);  float tx = sx - fx;
  int x0 = (int)fx, x1 = x0 + 1;
  x0 = x0 < 0 ? 0 : (x0 >= W ? W - 1 : x0);
  x1 = x1 < 0 ? 0 : (x1 >= W ? W - 1 : x1);
  float v00 = (float)p[y0 * W + x0], v01 = (float)p[y0 * W + x1];
  float v10 = (float)p[y1 * W + x0], v11 = (float)p[y1 * W + x1];
  float v0 = v00 + tx * (v01 - v00);
  float v1 = v10 + tx * (v11 - v10);
  return v0 + ty * (v1 - v0);
}

// ---------------- blocked WMMA GEMM with async double-buffered B -----------
// C[M,N] = act(A[M,K]_f16 * B^T + bias), B f16 [N,K] row-major.
// Wave tile 16Mx64N; block = 4 waves in M; B tile 64x32 f16 in LDS, double
// buffered via global_load_async_to_lds_b128 + s_wait_asynccnt.
// M % 64 == 0, K % 32 == 0, rows 16B-aligned. Cf/Ch optional dual outputs.
template<int ACT>  // 0 none, 1 relu
__global__ __launch_bounds__(128)
void gemm_wmma_h(const _Float16* __restrict__ A, const _Float16* __restrict__ B,
                 const float* __restrict__ bias,
                 float* __restrict__ Cf, _Float16* __restrict__ Ch,
                 int M, int N, int K)
{
  __shared__ __align__(16) _Float16 Bs[2][64][32];

  const int lane  = threadIdx.x & 31;
  const int wave  = threadIdx.x >> 5;
  const int tm    = blockIdx.x * 4 + wave;
  const int n0    = blockIdx.y * 64;
  const int row16 = lane & 15;
  const int half  = lane >> 4;
  const int m     = tm * 16 + row16;
  const _Float16* __restrict__ Arow = A + (size_t)m * K;

  // staging role: thread t owns B row (n0 + t/2), k-half (t&1)*16
  const int sn  = threadIdx.x >> 1;
  const int sk  = (threadIdx.x & 1) * 16;
  const int gn  = n0 + sn;
  const bool gok = (gn < N);
  const _Float16* __restrict__ Bg = B + (size_t)gn * K + sk;

  // pad rows (n >= N) zeroed once; async path never touches them
  const h8 hz = (h8)(_Float16)0.f;
  if (!gok) {
    *(h8*)&Bs[0][sn][sk] = hz; *(h8*)&Bs[0][sn][sk + 8] = hz;
    *(h8*)&Bs[1][sn][sk] = hz; *(h8*)&Bs[1][sn][sk + 8] = hz;
  } else {
    async_copy_b128(Bg,     &Bs[0][sn][sk]);
    async_copy_b128(Bg + 8, &Bs[0][sn][sk + 8]);
  }

  v8f acc[4];
#pragma unroll
  for (int j = 0; j < 4; ++j) acc[j] = (v8f){0.f,0.f,0.f,0.f,0.f,0.f,0.f,0.f};

  int buf = 0;
  for (int kb = 0; kb < K; kb += 32) {
    wait_async0();        // own async group for Bs[buf] has landed
    __syncthreads();      // all waves' groups landed -> tile complete
    // issue next tile into the other buffer (overlaps this tile's WMMAs)
    if (kb + 32 < K) {
      if (gok) {
        async_copy_b128(Bg + kb + 32, &Bs[buf ^ 1][sn][sk]);
        async_copy_b128(Bg + kb + 40, &Bs[buf ^ 1][sn][sk + 8]);
      }
      __builtin_prefetch(Arow + kb + 32, 0, 3);
    }

    // A fragment (ISA 7.12.2: lane-half selects K sub-range)
    h8 a0 = *(const h8*)(Arow + kb + half * 16);
    h8 a1 = *(const h8*)(Arow + kb + 8 + half * 16);
    v16h a = __builtin_shufflevector(a0, a1, 0,1,2,3,4,5,6,7,8,9,10,11,12,13,14,15);

#pragma unroll
    for (int j = 0; j < 4; ++j) {
      const h8* bp = (const h8*)&Bs[buf][j * 16 + row16][half * 16];
      h8 b0 = bp[0], b1 = bp[1];
      v16h b = __builtin_shufflevector(b0, b1, 0,1,2,3,4,5,6,7,8,9,10,11,12,13,14,15);
      acc[j] = __builtin_amdgcn_wmma_f32_16x16x32_f16(false, a, false, b,
                                                      (short)0, acc[j], false, false);
    }
    buf ^= 1;
  }

  // epilogue: C element r -> M = r + 8*half, N = lane&15
#pragma unroll
  for (int j = 0; j < 4; ++j) {
    int n = n0 + j * 16 + row16;
    if (n < N) {
      float bv = bias ? bias[n] : 0.0f;
#pragma unroll
      for (int r = 0; r < 8; ++r) {
        int mr = tm * 16 + r + half * 8;
        float v = acc[j][r] + bv;
        if (ACT == 1) v = fmaxf(v, 0.0f);
        if (Cf) Cf[(size_t)mr * N + n] = v;
        if (Ch) Ch[(size_t)mr * N + n] = (_Float16)v;
      }
    }
  }
}

// ---------------- implicit-GEMM convolution on WMMA ------------------------
// A gathered (im2col) from f16 input; B = f16 weights [Cout, KP] (K-padded),
// async double-buffered in LDS.  UP2: input is bilinear 2x upsample of `in`.
template<int UP2>
__device__ __forceinline__ float conv_fetchA(const _Float16* __restrict__ in,
    int b, int oy, int ox, int k, int Cin, int Hin, int Win,
    int KH, int KW, int stride, int pad)
{
  const int khw = KH * KW;
  int ci = k / khw;  int r = k - ci * khw;
  int ky = r / KW;   int kx = r - ky * KW;
  int iy = oy * stride - pad + ky;
  int ix = ox * stride - pad + kx;
  const int He = UP2 ? 2 * Hin : Hin;
  const int We = UP2 ? 2 * Win : Win;
  if (iy < 0 || iy >= He || ix < 0 || ix >= We) return 0.0f;
  const _Float16* ic = in + ((size_t)b * Cin + ci) * Hin * Win;
  return UP2 ? up2_fetch(ic, Hin, Win, iy, ix) : (float)ic[iy * Win + ix];
}

template<int UP2, int ACT>  // ACT: 1 relu, 2 sigmoid
__global__ __launch_bounds__(128)
void conv_wmma_h(const _Float16* __restrict__ in, const _Float16* __restrict__ w,
                 const float* __restrict__ bias,
                 float* __restrict__ outf, _Float16* __restrict__ outh,
                 int Bn, int Cin, int Hin, int Win,
                 int Cout, int Hout, int Wout,
                 int KH, int KW, int stride, int pad, int KP)
{
  __shared__ __align__(16) _Float16 Bs[2][64][32];

  const int Kreal = Cin * KH * KW;
  const int lane  = threadIdx.x & 31;
  const int wave  = threadIdx.x >> 5;
  const int tm    = blockIdx.x * 4 + wave;
  const int n0    = blockIdx.y * 64;
  const int row16 = lane & 15;
  const int half  = lane >> 4;
  const int hw    = Hout * Wout;

  const int m  = tm * 16 + row16;
  const int b  = m / hw;
  const int ps = m - b * hw;
  const int oy = ps / Wout;
  const int ox = ps - oy * Wout;

  const int sn  = threadIdx.x >> 1;
  const int sk  = (threadIdx.x & 1) * 16;
  const int gn  = n0 + sn;
  const bool gok = (gn < Cout);
  const _Float16* __restrict__ Bg = w + (size_t)gn * KP + sk;

  const h8 hz = (h8)(_Float16)0.f;
  if (!gok) {
    *(h8*)&Bs[0][sn][sk] = hz; *(h8*)&Bs[0][sn][sk + 8] = hz;
    *(h8*)&Bs[1][sn][sk] = hz; *(h8*)&Bs[1][sn][sk + 8] = hz;
  } else {
    async_copy_b128(Bg,     &Bs[0][sn][sk]);
    async_copy_b128(Bg + 8, &Bs[0][sn][sk + 8]);
  }

  v8f acc[4];
#pragma unroll
  for (int j = 0; j < 4; ++j) acc[j] = (v8f){0.f,0.f,0.f,0.f,0.f,0.f,0.f,0.f};

  int buf = 0;
  for (int kb = 0; kb < KP; kb += 32) {
    wait_async0();
    __syncthreads();
    if (kb + 32 < KP && gok) {
      async_copy_b128(Bg + kb + 32, &Bs[buf ^ 1][sn][sk]);
      async_copy_b128(Bg + kb + 40, &Bs[buf ^ 1][sn][sk + 8]);
    }

    // A fragment (gathered im2col)
    v16h a;
    const int ka0 = kb + half * 16;
    const int ka1 = kb + 8 + half * 16;
#pragma unroll
    for (int e = 0; e < 8; ++e) {
      int k0 = ka0 + e, k1 = ka1 + e;
      a[e]     = (_Float16)((k0 < Kreal) ? conv_fetchA<UP2>(in, b, oy, ox, k0,
                                 Cin, Hin, Win, KH, KW, stride, pad) : 0.0f);
      a[e + 8] = (_Float16)((k1 < Kreal) ? conv_fetchA<UP2>(in, b, oy, ox, k1,
                                 Cin, Hin, Win, KH, KW, stride, pad) : 0.0f);
    }

#pragma unroll
    for (int j = 0; j < 4; ++j) {
      const h8* bp = (const h8*)&Bs[buf][j * 16 + row16][half * 16];
      h8 b0 = bp[0], b1 = bp[1];
      v16h bfr = __builtin_shufflevector(b0, b1, 0,1,2,3,4,5,6,7,8,9,10,11,12,13,14,15);
      acc[j] = __builtin_amdgcn_wmma_f32_16x16x32_f16(false, a, false, bfr,
                                                      (short)0, acc[j], false, false);
    }
    buf ^= 1;
  }

#pragma unroll
  for (int j = 0; j < 4; ++j) {
    int n = n0 + j * 16 + row16;
    if (n < Cout) {
      float bv = bias[n];
#pragma unroll
      for (int r = 0; r < 8; ++r) {
        int mr  = tm * 16 + r + half * 8;
        int b2  = mr / hw;
        int p2  = mr - b2 * hw;
        int oy2 = p2 / Wout;
        int ox2 = p2 - oy2 * Wout;
        float v = acc[j][r] + bv;
        if (ACT == 1)      v = fmaxf(v, 0.0f);
        else if (ACT == 2) v = 1.0f / (1.0f + __expf(-v));
        size_t o = (((size_t)b2 * Cout + n) * Hout + oy2) * Wout + ox2;
        if (outf) outf[o] = v;
        if (outh) outh[o] = (_Float16)v;
      }
    }
  }
}

// ---------------- row squared-norm: out[r] = sum_d X[r,d]^2 ----------------
__global__ void rownorm2_kernel(const float* __restrict__ X,
                                float* __restrict__ out, int D)
{
  __shared__ float red[256];
  const int r = blockIdx.x, t = threadIdx.x;
  float v = 0.0f;
  for (int d = t; d < D; d += 256) {
    float x = X[(size_t)r * D + d];
    v += x * x;
  }
  red[t] = v; __syncthreads();
  for (int s = 128; s > 0; s >>= 1) {
    if (t < s) red[t] += red[t + s];
    __syncthreads();
  }
  if (t == 0) out[r] = red[0];
}

// ---------------- SoftSOM softmax over prototypes --------------------------
__global__ void som_softmax_kernel(float* __restrict__ zp,
                                   const float* __restrict__ zn2,
                                   const float* __restrict__ pn2,
                                   _Float16* __restrict__ wh,
                                   int Kn, int KP)
{
  __shared__ float red[256];
  const int b = blockIdx.x, t = threadIdx.x;
  float* row = zp + (size_t)b * Kn;
  const float zz = zn2[b];

  float mx = -3.402823e38f;
  for (int j = t; j < Kn; j += 256) {
    float d2 = zz + pn2[j] - 2.0f * row[j];
    mx = fmaxf(mx, -sqrtf(fmaxf(d2, 0.0f)) * INV_TEMP);
  }
  red[t] = mx; __syncthreads();
  for (int s = 128; s > 0; s >>= 1) {
    if (t < s) red[t] = fmaxf(red[t], red[t + s]);
    __syncthreads();
  }
  mx = red[0]; __syncthreads();

  float sum = 0.0f;
  for (int j = t; j < Kn; j += 256) {
    float d2 = zz + pn2[j] - 2.0f * row[j];
    float e  = __expf(-sqrtf(fmaxf(d2, 0.0f)) * INV_TEMP - mx);
    row[j] = e;
    sum += e;
  }
  red[t] = sum; __syncthreads();
  for (int s = 128; s > 0; s >>= 1) {
    if (t < s) red[t] += red[t + s];
    __syncthreads();
  }
  const float inv = 1.0f / red[0];
  for (int j = t; j < KP; j += 256)
    wh[(size_t)b * KP + j] = (_Float16)((j < Kn) ? row[j] * inv : 0.0f);
}

// ---------------- z_in = LayerNorm(z0 + 0.5*blended), D = 256 --------------
__global__ void ln_fuse_kernel(const float* __restrict__ z0,
                               const float* __restrict__ bl,
                               float* __restrict__ outf,
                               _Float16* __restrict__ outh)
{
  __shared__ float red[256];
  const int b = blockIdx.x, t = threadIdx.x;
  float v = z0[(size_t)b * DLAT + t] + 0.5f * bl[(size_t)b * DLAT + t];
  red[t] = v; __syncthreads();
  for (int s = 128; s > 0; s >>= 1) {
    if (t < s) red[t] += red[t + s];
    __syncthreads();
  }
  const float mean = red[0] * (1.0f / 256.0f); __syncthreads();
  const float d = v - mean;
  red[t] = d * d; __syncthreads();
  for (int s = 128; s > 0; s >>= 1) {
    if (t < s) red[t] += red[t + s];
    __syncthreads();
  }
  const float var = red[0] * (1.0f / 256.0f);
  float o = d * rsqrtf(var + 1e-5f);
  outf[(size_t)b * DLAT + t] = o;
  outh[(size_t)b * DLAT + t] = (_Float16)o;
}

// ---------------- attention weights (tiny, one block of 1024) --------------
__global__ void attn_kernel(const float* __restrict__ ce,
                            const float* __restrict__ ne,
                            const float* __restrict__ qw, const float* __restrict__ qb,
                            const float* __restrict__ aw, const float* __restrict__ ab,
                            float* __restrict__ attn_w)
{
  __shared__ float qmean[256];
  __shared__ float q[256];
  __shared__ float proj[4][1024];
  __shared__ float lg[16];
  const int t = threadIdx.x;

  if (t < 256) {
    float s = 0.0f;
    for (int c = 0; c < 10; ++c) s += ce[c * 256 + t];
    qmean[t] = s * 0.1f;
  }
  __syncthreads();
  if (t < 256) {
    float s = 0.0f;
    for (int e = 0; e < 256; ++e) s += qmean[e] * qw[t * 256 + e];
    q[t] = s + qb[t];
  }
  for (int n = 0; n < 4; ++n) {
    float s = 0.0f;
    for (int e = 0; e < 256; ++e) s += ne[n * 256 + e] * aw[(size_t)t * 256 + e];
    proj[n][t] = s + ab[t];
  }
  __syncthreads();
  if (t < 16) {
    int h = t >> 2, n = t & 3;
    float s = 0.0f;
    for (int d = 0; d < 256; ++d) s += q[d] * proj[n][h * 256 + d];
    lg[h * 4 + n] = s;
  }
  __syncthreads();
  if (t < 4) {
    float m = fmaxf(fmaxf(lg[t*4+0], lg[t*4+1]), fmaxf(lg[t*4+2], lg[t*4+3]));
    float e0 = __expf(lg[t*4+0]-m), e1 = __expf(lg[t*4+1]-m);
    float e2 = __expf(lg[t*4+2]-m), e3 = __expf(lg[t*4+3]-m);
    float s = e0 + e1 + e2 + e3;
    attn_w[t*4+0] = e0/s; attn_w[t*4+1] = e1/s;
    attn_w[t*4+2] = e2/s; attn_w[t*4+3] = e3/s;
  }
}

// ---------------- fused[b, h*256+d] = sum_n attn_w[h,n]*topo[n,b,d] --------
__global__ void fuse_kernel(const float* __restrict__ topo,
                            const float* __restrict__ aw,
                            float* __restrict__ fused,
                            _Float16* __restrict__ fusedh)
{
  __shared__ float w[16];
  if (threadIdx.x < 16) w[threadIdx.x] = aw[threadIdx.x];
  __syncthreads();
  const size_t idx = (size_t)blockIdx.x * 256 + threadIdx.x;
  const int b = (int)(idx >> 10);
  const int o = (int)(idx & 1023);
  const int h = o >> 8, d = o & 255;
  float s = 0.0f;
#pragma unroll
  for (int n = 0; n < 4; ++n)
    s += w[h * 4 + n] * topo[((size_t)n * BATCH + b) * DLAT + d];
  fused[idx]  = s;
  fusedh[idx] = (_Float16)s;
}

// ---------------------------------------------------------------------------
extern "C" void kernel_launch(void* const* d_in, const int* in_sizes, int n_in,
                              void* d_out, int out_size, void* d_ws, size_t ws_size,
                              hipStream_t stream) {
  (void)in_sizes; (void)n_in; (void)out_size; (void)ws_size;

  const float* x     = (const float*)d_in[0];
  const float* ew1   = (const float*)d_in[1];
  const float* eb1   = (const float*)d_in[2];
  const float* ew2   = (const float*)d_in[3];
  const float* eb2   = (const float*)d_in[4];
  const float* ew3   = (const float*)d_in[5];
  const float* eb3   = (const float*)d_in[6];
  const float* ew4   = (const float*)d_in[7];
  const float* eb4   = (const float*)d_in[8];
  const float* efc_w = (const float*)d_in[9];
  const float* efc_b = (const float*)d_in[10];
  const float* dfc_w = (const float*)d_in[11];
  const float* dfc_b = (const float*)d_in[12];
  const float* dw1   = (const float*)d_in[13];
  const float* db1   = (const float*)d_in[14];
  const float* dw2   = (const float*)d_in[15];
  const float* db2   = (const float*)d_in[16];
  const float* nfw   = (const float*)d_in[17];
  const float* nfb   = (const float*)d_in[18];
  const float* protos[4] = {(const float*)d_in[19], (const float*)d_in[20],
                            (const float*)d_in[21], (const float*)d_in[22]};
  const float* ce    = (const float*)d_in[23];
  const float* nem   = (const float*)d_in[24];
  const float* qw    = (const float*)d_in[25];
  const float* qb    = (const float*)d_in[26];
  const float* aw    = (const float*)d_in[27];
  const float* ab    = (const float*)d_in[28];
  const float* clsw  = (const float*)d_in[29];
  const float* clsb  = (const float*)d_in[30];

  float* out_logits = (float*)d_out;                          // [2048,10]
  float* out_fused  = out_logits + (size_t)BATCH * 10;        // [2048,1024]
  float* out_topo   = out_fused + (size_t)BATCH * 4 * DLAT;   // [4,2048,256]
  float* out_recon  = out_topo + (size_t)4 * BATCH * DLAT;    // [2048,3,32,32]

  const int KN[4] = {6400, 1600, 400, 100};
  const int KQ[4] = {6400, 1600, 416, 128};   // K padded to mult of 32

  // -------- workspace bump allocation (deterministic; ~200 MB peak) --------
  char* p = (char*)d_ws;
  auto ah = [&](size_t n) -> _Float16* {
    _Float16* r = (_Float16*)p; p += (n * 2 + 255) & ~(size_t)255; return r; };
  auto af = [&](size_t n) -> float* {
    float* r = (float*)p; p += (n * 4 + 255) & ~(size_t)255; return r; };

  _Float16* ew1h = ah((size_t)32 * 96);
  _Float16* ew2h = ah((size_t)64 * 288);
  _Float16* ew3h = ah((size_t)128 * 576);
  _Float16* ew4h = ah((size_t)256 * 1152);
  _Float16* dw1h = ah((size_t)64 * 1152);
  _Float16* dw2h = ah((size_t)3 * 576);
  _Float16* efch = ah((size_t)256 * 16384);
  _Float16* dfch = ah((size_t)8192 * 256);
  _Float16* nfwh = ah((size_t)1024 * 256);
  _Float16* clsh = ah((size_t)10 * 1024);
  _Float16* ph[4];  for (int i = 0; i < 4; ++i) ph[i]  = ah((size_t)KN[i] * 256);
  _Float16* pth[4]; for (int i = 0; i < 4; ++i) pth[i] = ah((size_t)256 * KQ[i]);
  _Float16* xh  = ah((size_t)BATCH * 3 * 1024);
  _Float16* h1h = ah((size_t)BATCH * 32 * 225);
  _Float16* h2h = ah((size_t)BATCH * 64 * 64);
  _Float16* h3h = ah((size_t)BATCH * 128 * 64);
  _Float16* h4h = ah((size_t)BATCH * 256 * 64);
  float* z0  = af((size_t)BATCH * DLAT);
  float* zb  = af((size_t)BATCH * DLAT);
  float* bl  = af((size_t)BATCH * DLAT);
  float* zn  = af(BATCH);
  float* pn  = af(8500);
  float* at  = af(16);
  _Float16* z0h   = ah((size_t)BATCH * DLAT);
  _Float16* zbh   = ah((size_t)BATCH * DLAT);
  _Float16* topoh = ah((size_t)4 * BATCH * DLAT);
  _Float16* fusedh= ah((size_t)BATCH * 1024);
  // lifetime-reused aliases (stream-serialized):
  _Float16* dpre_h = h1h;                          // [B,8192]  in h1+h2
  _Float16* g_h    = h4h;                          // [B,64,16,16] in h4 (exact)
  float*    zp     = (float*)h1h;                  // [B,<=6400] in h1..h3
  _Float16* wh     = (_Float16*)((char*)h1h + (size_t)BATCH * 6400 * 4);

  const dim3 blk(128);
  const size_t pnoff[4] = {0, 6400, 8000, 8400};

  // -------- one-time f32 -> f16 conversions --------
  auto cvt = [&](const float* s, _Float16* d, int K, int KP, size_t rows) {
    size_t tot = rows * (size_t)KP;
    cvt_pad_kernel<<<(unsigned)((tot + 255) / 256), 256, 0, stream>>>(s, d, K, KP, tot);
  };
  cvt(ew1, ew1h, 75, 96, 32);
  cvt(ew2, ew2h, 288, 288, 64);
  cvt(ew3, ew3h, 576, 576, 128);
  cvt(ew4, ew4h, 1152, 1152, 256);
  cvt(dw1, dw1h, 1152, 1152, 64);
  cvt(dw2, dw2h, 576, 576, 3);
  cvt(efc_w, efch, 16384, 16384, 256);
  cvt(dfc_w, dfch, 256, 256, 8192);
  cvt(nfw, nfwh, 256, 256, 1024);
  cvt(clsw, clsh, 1024, 1024, 10);
  for (int i = 0; i < 4; ++i) cvt(protos[i], ph[i], 256, 256, KN[i]);
  for (int i = 0; i < 4; ++i) {
    size_t tot = (size_t)256 * KQ[i];
    transpose_pad_kernel<<<(unsigned)((tot + 255) / 256), 256, 0, stream>>>(
        protos[i], pth[i], KN[i], KQ[i]);
  }
  cvt(x, xh, 1024, 1024, (size_t)BATCH * 3);
  for (int i = 0; i < 4; ++i)
    rownorm2_kernel<<<KN[i], 256, 0, stream>>>(protos[i], pn + pnoff[i], DLAT);

  // -------- encoder (implicit-GEMM convs on WMMA, f16 activations) --------
  { int M = BATCH * 225;  dim3 g(M / 64, 1);
    conv_wmma_h<0,1><<<g, blk, 0, stream>>>(xh, ew1h, eb1, nullptr, h1h,
        BATCH, 3, 32, 32, 32, 15, 15, 5, 5, 2, 1, 96); }
  { int M = BATCH * 64;   dim3 g(M / 64, 1);
    conv_wmma_h<0,1><<<g, blk, 0, stream>>>(h1h, ew2h, eb2, nullptr, h2h,
        BATCH, 32, 15, 15, 64, 8, 8, 3, 3, 2, 1, 288); }
  { int M = BATCH * 64;   dim3 g(M / 64, 2);
    conv_wmma_h<0,1><<<g, blk, 0, stream>>>(h2h, ew3h, eb3, nullptr, h3h,
        BATCH, 64, 8, 8, 128, 8, 8, 3, 3, 1, 1, 576); }
  { int M = BATCH * 64;   dim3 g(M / 64, 4);
    conv_wmma_h<0,1><<<g, blk, 0, stream>>>(h3h, ew4h, eb4, nullptr, h4h,
        BATCH, 128, 8, 8, 256, 8, 8, 3, 3, 1, 1, 1152); }
  { dim3 g(BATCH / 64, 4);
    gemm_wmma_h<0><<<g, blk, 0, stream>>>(h4h, efch, efc_b, z0, z0h,
        BATCH, 256, 16384); }

  // -------- decoder --------
  { dim3 g(BATCH / 64, 128);
    gemm_wmma_h<1><<<g, blk, 0, stream>>>(z0h, dfch, dfc_b, nullptr, dpre_h,
        BATCH, 8192, 256); }
  { int M = BATCH * 256;  dim3 g(M / 64, 1);
    conv_wmma_h<1,1><<<g, blk, 0, stream>>>(dpre_h, dw1h, db1, nullptr, g_h,
        BATCH, 128, 8, 8, 64, 16, 16, 3, 3, 1, 1, 1152); }
  { int M = BATCH * 1024; dim3 g(M / 64, 1);
    conv_wmma_h<1,2><<<g, blk, 0, stream>>>(g_h, dw2h, db2, out_recon, nullptr,
        BATCH, 64, 16, 16, 3, 32, 32, 3, 3, 1, 1, 576); }

  // -------- hierarchical SoftSOM chain --------
  const _Float16* zin_h = z0h;
  for (int i = 0; i < 4; ++i) {
    { dim3 g(BATCH / 64, 4);
      gemm_wmma_h<0><<<g, blk, 0, stream>>>(zin_h, nfwh + (size_t)i * DLAT * DLAT,
          nfb + (size_t)i * DLAT, zb, zbh, BATCH, 256, 256); }
    rownorm2_kernel<<<BATCH, 256, 0, stream>>>(zb, zn, DLAT);
    { dim3 g(BATCH / 64, (KN[i] + 63) / 64);
      gemm_wmma_h<0><<<g, blk, 0, stream>>>(zbh, ph[i], nullptr, zp, nullptr,
          BATCH, KN[i], 256); }
    som_softmax_kernel<<<BATCH, 256, 0, stream>>>(zp, zn, pn + pnoff[i], wh,
        KN[i], KQ[i]);
    { dim3 g(BATCH / 64, 4);
      gemm_wmma_h<0><<<g, blk, 0, stream>>>(wh, pth[i], nullptr, bl, nullptr,
          BATCH, 256, KQ[i]); }
    float* topo_i = out_topo + (size_t)i * BATCH * DLAT;
    _Float16* topoh_i = topoh + (size_t)i * BATCH * DLAT;
    ln_fuse_kernel<<<BATCH, 256, 0, stream>>>(z0, bl, topo_i, topoh_i);
    zin_h = topoh_i;
  }

  // -------- attention fusion + classifier --------
  attn_kernel<<<1, 1024, 0, stream>>>(ce, nem, qw, qb, aw, ab, at);
  fuse_kernel<<<(BATCH * 1024) / 256, 256, 0, stream>>>(out_topo, at,
      out_fused, fusedh);
  { dim3 g(BATCH / 64, 1);
    gemm_wmma_h<0><<<g, blk, 0, stream>>>(fusedh, clsh, clsb, out_logits, nullptr,
        BATCH, 10, 1024); }
}
